// DisplacementDecoder_33818572489151
// MI455X (gfx1250) — matrix operand order repaired and checked by reading, not compiled, not run on hardware
//
#include <hip/hip_runtime.h>
#include <math.h>

// ---------------- problem constants (match reference) ----------------
#define B_   32
#define P_   16384
#define C_   2
#define S_   256
#define LAT  10
#define ENC  10
#define H_   64
#define NL   4
#define DIN  70          // 6*ENC + LAT
#define KPAD 96          // input features padded to 3 WMMA K-steps of 32

typedef __attribute__((ext_vector_type(16))) __bf16 v16bf;
typedef __attribute__((ext_vector_type(8)))  float  v8f;
typedef unsigned int v4u __attribute__((ext_vector_type(4)));
typedef int          v8i __attribute__((ext_vector_type(8)));
typedef int          v4i __attribute__((ext_vector_type(4)));

#ifndef __has_builtin
#define __has_builtin(x) 0
#endif

#if __has_builtin(__builtin_amdgcn_tensor_load_to_lds) && \
    __has_builtin(__builtin_amdgcn_s_wait_tensorcnt)
#define USE_TDM 1
#else
#define USE_TDM 0
#endif

union Frag { v16bf v; unsigned u[8]; };

// A fragment (16x32 bf16): lane holds row M=lane&15; VGPR v holds K pair at
// K = 2*(v&3) + 8*(lane>=16) + 16*(v>=4)  (+ koff)
__device__ __forceinline__ v16bf load_a_frag(const __bf16* xbase, int lane, int koff) {
  Frag a;
  const int row = lane & 15, g = lane >> 4;
  const __bf16* rp = xbase + row * KPAD + koff + 8 * g;
#pragma unroll
  for (int v = 0; v < 8; ++v)
    a.u[v] = *(const unsigned*)(rp + 2 * (v & 3) + 16 * (v >> 2));
  return a.v;
}

// B fragment (32x16 bf16): lane holds col N=(lane&15)+ncol; VGPR v holds K pair
// q = K/2 = qbase + 4*(lane>=16) + (v&3) + 8*(v>=4). Weights stored pre-packed: (q, n) -> 2 halves.
__device__ __forceinline__ v16bf load_b_frag(const __bf16* wq, int ncols, int qbase,
                                             int ncol, int lane) {
  Frag b;
  const int g = lane >> 4;
  const int n = (lane & 15) + ncol;
#pragma unroll
  for (int v = 0; v < 8; ++v) {
    int q = qbase + 4 * g + (v & 3) + 8 * (v >> 2);
    b.u[v] = *(const unsigned*)(wq + (q * ncols + n) * 2);
  }
  return b.v;
}

__device__ __forceinline__ float elu(float x) {
  return x > 0.f ? x : (__expf(x) - 1.f);
}

// ------------------------------------------------------------------
// Kernel 1: fused MLP (WMMA) + head + projection + bilinear splat
// grid 512 blocks x 256 threads (8 waves); each wave does 8 tiles of 16 rows
// ------------------------------------------------------------------
__global__ void __launch_bounds__(256)
k_mlp_splat(const float* __restrict__ z, const float* __restrict__ ori,
            const float* __restrict__ mpos,
            const float* __restrict__ Win, const float* __restrict__ Wl,
            const float* __restrict__ Wo1, const float* __restrict__ Wo2,
            const float* __restrict__ amp, const float* __restrict__ ampvar,
            float* __restrict__ imgR,
            float* __restrict__ out_final, float* __restrict__ out_disp)
{
  __shared__ __bf16 s_win[48 * 64 * 2];        // W_in  padded to K=96, packed K-pairs
  __shared__ __bf16 s_wl[NL][32 * 64 * 2];     // 4 hidden layers 64x64
  __shared__ __bf16 s_wo[32 * 16 * 2];         // W_out1 64x3 padded to N=16
  __shared__ float  s_w2[9];
  __shared__ float  s_freq[ENC];
  __shared__ __bf16 s_x[8][16 * KPAD];         // per-wave activation staging
  __shared__ float  s_g[8][16][4];             // per-wave head output (3 cols)
#if USE_TDM
  __shared__ float  s_wl_raw[NL * H_ * H_];    // raw f32 W_layers, filled by TDM
#endif

  const int tid  = threadIdx.x;
  const int lane = tid & 31;
  const int w    = tid >> 5;

#if USE_TDM
  // ---- async Tensor-DMA: pull all of W_layers (64 KB f32) into LDS ----
  if (tid == 0) {
    unsigned lds_off = (unsigned)(unsigned long long)(uintptr_t)(void*)s_wl_raw;
    unsigned long long ga = (unsigned long long)(uintptr_t)(const void*)Wl;
    // D# group0: count=1 | lds_addr | global_addr[56:0] | type=2
    v4u g0 = { 1u, lds_off,
               (unsigned)(ga & 0xffffffffull),
               (unsigned)((ga >> 32) & 0x01ffffffull) | 0x80000000u };
    // D# group1: data_size=4B; 1-row 2D tensor: dim0=tile0=16384, dim1=tile1=1
    v8i g1 = { (int)0x20000,      // wg_mask=0, data_size=2 (4 bytes)
               (int)0x40000000,   // tensor_dim0[15:0]=0x4000 in bits[63:48]
               (int)0x10000,      // tensor_dim0 hi=0, tensor_dim1 lo=1
               (int)0x40000000,   // tensor_dim1 hi=0, tile_dim0=0x4000
               1,                 // tile_dim1=1, tile_dim2=0
               16384,             // tensor_dim0_stride lo
               0, 0 };
    v4i gz = { 0, 0, 0, 0 };
#if __clang_major__ >= 23
    v8i gz8 = { 0, 0, 0, 0, 0, 0, 0, 0 };
    __builtin_amdgcn_tensor_load_to_lds(g0, g1, gz, gz, gz8, 0);
#else
    __builtin_amdgcn_tensor_load_to_lds(g0, g1, gz, gz, 0);
#endif
    __builtin_amdgcn_s_wait_tensorcnt(0);
  }
#endif

  // ---- stage small weights into LDS (bf16, K-pair packed) ----
  for (int i = tid; i < 48 * 64; i += 256) {
    int q = i >> 6, n = i & 63;
    int k0 = 2 * q, k1 = 2 * q + 1;
    s_win[(i << 1) + 0] = (__bf16)(k0 < DIN ? Win[k0 * H_ + n] : 0.f);
    s_win[(i << 1) + 1] = (__bf16)(k1 < DIN ? Win[k1 * H_ + n] : 0.f);
  }
  for (int i = tid; i < 32 * 16; i += 256) {
    int q = i >> 4, n = i & 15;
    s_wo[(i << 1) + 0] = (__bf16)(n < 3 ? Wo1[(2 * q) * 3 + n] : 0.f);
    s_wo[(i << 1) + 1] = (__bf16)(n < 3 ? Wo1[(2 * q + 1) * 3 + n] : 0.f);
  }
  if (tid < 9)   s_w2[tid]   = Wo2[tid];
  if (tid < ENC) s_freq[tid] = __powf((float)(S_ / 2), (float)tid / (float)(ENC - 1));

#if USE_TDM
  __syncthreads();   // TDM data now visible (issuing wave waited on TENSORcnt)
  for (int i = tid; i < NL * 32 * 64; i += 256) {
    int l = i >> 11, r = i & 2047;
    int q = r >> 6, n = r & 63;
    const float* Wb = s_wl_raw + l * H_ * H_;
    s_wl[l][(r << 1) + 0] = (__bf16)Wb[(2 * q) * H_ + n];
    s_wl[l][(r << 1) + 1] = (__bf16)Wb[(2 * q + 1) * H_ + n];
  }
#else
  for (int i = tid; i < NL * 32 * 64; i += 256) {
    int l = i >> 11, r = i & 2047;
    int q = r >> 6, n = r & 63;
    const float* Wb = Wl + l * H_ * H_;
    s_wl[l][(r << 1) + 0] = (__bf16)Wb[(2 * q) * H_ + n];
    s_wl[l][(r << 1) + 1] = (__bf16)Wb[(2 * q + 1) * H_ + n];
  }
#endif
  __syncthreads();

  const int gw = blockIdx.x * 8 + w;           // global wave id, 4096 waves
  const float twopi = 6.28318530718f;

  for (int it = 0; it < 8; ++it) {
    const int tile = gw * 8 + it;              // 32768 tiles of 16 rows
    const int b  = tile >> 10;                 // 1024 tiles per batch (P/16)
    const int p0 = (tile & 1023) << 4;

    // prefetch next tile's positions (global_prefetch_b8)
    if (it + 1 < 8) {
      const int tn = tile + 1;
      __builtin_prefetch(mpos + (size_t)((tn & 1023) << 4) * 3, 0, 0);
    }

    // ---- stage x = [posenc(60) | z(10) | pad] as bf16 into per-wave LDS ----
    // branch-free: one __sincosf per (row, coord, freq) fills both halves
    __bf16* xw = s_x[w];
    for (int i = lane; i < 16 * 30; i += 32) {
      int row = i / 30, j = i - row * 30;      // j = coord*10 + f
      int coord = j / 10, f = j - coord * 10;
      float pos = mpos[(p0 + row) * 3 + coord];
      float k = twopi * pos * s_freq[f];
      float sv, cv;
      __sincosf(k, &sv, &cv);
      int cb = coord * 20 + f;
      xw[row * KPAD + cb]      = (__bf16)sv;
      xw[row * KPAD + cb + 10] = (__bf16)cv;
    }
    for (int i = lane; i < 16 * 36; i += 32) { // cols 60..95: z then zero pad
      int row = i / 36, col = 60 + (i - row * 36);
      float val = (col < DIN) ? z[b * LAT + (col - 60)] : 0.f;
      xw[row * KPAD + col] = (__bf16)val;
    }

    const int g16 = lane >> 4;
    const int ncl = lane & 15;

    // ---- input layer: [16,96] @ [96,64], no activation ----
    {
      v16bf a0 = load_a_frag(xw, lane, 0);
      v16bf a1 = load_a_frag(xw, lane, 32);
      v16bf a2 = load_a_frag(xw, lane, 64);
#pragma unroll
      for (int t = 0; t < 4; ++t) {
        v8f c = {};
        c = __builtin_amdgcn_wmma_f32_16x16x32_bf16(false, a0, false,
              load_b_frag(s_win, 64, 0,  16 * t, lane), (short)0, c, false, false);
        c = __builtin_amdgcn_wmma_f32_16x16x32_bf16(false, a1, false,
              load_b_frag(s_win, 64, 16, 16 * t, lane), (short)0, c, false, false);
        c = __builtin_amdgcn_wmma_f32_16x16x32_bf16(false, a2, false,
              load_b_frag(s_win, 64, 32, 16 * t, lane), (short)0, c, false, false);
#pragma unroll
        for (int v = 0; v < 8; ++v)
          xw[(v + 8 * g16) * KPAD + (16 * t + ncl)] = (__bf16)c[v];
      }
    }

    // ---- hidden layers: h = elu(h @ W) ----
#pragma unroll
    for (int l = 0; l < NL; ++l) {
      v16bf a0 = load_a_frag(xw, lane, 0);
      v16bf a1 = load_a_frag(xw, lane, 32);
#pragma unroll
      for (int t = 0; t < 4; ++t) {
        v8f c = {};
        c = __builtin_amdgcn_wmma_f32_16x16x32_bf16(false, a0, false,
              load_b_frag(s_wl[l], 64, 0,  16 * t, lane), (short)0, c, false, false);
        c = __builtin_amdgcn_wmma_f32_16x16x32_bf16(false, a1, false,
              load_b_frag(s_wl[l], 64, 16, 16 * t, lane), (short)0, c, false, false);
#pragma unroll
        for (int v = 0; v < 8; ++v)
          xw[(v + 8 * g16) * KPAD + (16 * t + ncl)] = (__bf16)elu(c[v]);
      }
    }

    // ---- head: g = elu(h @ W_out1) (cols 0..2 valid) ----
    {
      v16bf a0 = load_a_frag(xw, lane, 0);
      v16bf a1 = load_a_frag(xw, lane, 32);
      v8f c = {};
      c = __builtin_amdgcn_wmma_f32_16x16x32_bf16(false, a0, false,
            load_b_frag(s_wo, 16, 0,  0, lane), (short)0, c, false, false);
      c = __builtin_amdgcn_wmma_f32_16x16x32_bf16(false, a1, false,
            load_b_frag(s_wo, 16, 16, 0, lane), (short)0, c, false, false);
      if (ncl < 3) {
#pragma unroll
        for (int v = 0; v < 8; ++v)
          s_g[w][v + 8 * g16][ncl] = elu(c[v]);
      }
    }

    // ---- finalize: @W_out2, + model_positions, project, splat ----
    if (lane < 16) {
      const int row = lane;
      const int p = p0 + row;
      float g0 = s_g[w][row][0], g1 = s_g[w][row][1], g2 = s_g[w][row][2];
      float d0 = g0 * s_w2[0] + g1 * s_w2[3] + g2 * s_w2[6];
      float d1 = g0 * s_w2[1] + g1 * s_w2[4] + g2 * s_w2[7];
      float d2 = g0 * s_w2[2] + g1 * s_w2[5] + g2 * s_w2[8];
      float m0 = mpos[p * 3 + 0], m1 = mpos[p * 3 + 1], m2 = mpos[p * 3 + 2];
      float f0 = m0 + d0, f1 = m1 + d1, f2 = m2 + d2;
      size_t o = (size_t)(b * P_ + p) * 3;
      out_disp[o + 0] = d0; out_disp[o + 1] = d1; out_disp[o + 2] = d2;
      out_final[o + 0] = f0; out_final[o + 1] = f1; out_final[o + 2] = f2;

      const float* R = ori + b * 9;
      float pr0 = R[0] * f0 + R[1] * f1 + R[2] * f2;
      float pr1 = R[3] * f0 + R[4] * f1 + R[5] * f2;

      float px = fminf(fmaxf((pr0 + 0.5f) * S_, 0.f), (float)S_ - 1.001f);
      float py = fminf(fmaxf((pr1 + 0.5f) * S_, 0.f), (float)S_ - 1.001f);
      int x0 = (int)floorf(px), y0 = (int)floorf(py);
      float fx = px - x0, fy = py - y0;
      int x1 = x0 + 1 < S_ - 1 ? x0 + 1 : S_ - 1;
      int y1 = y0 + 1 < S_ - 1 ? y0 + 1 : S_ - 1;

      float av0 = ampvar[p], av1 = ampvar[P_ + p];
      float mx = fmaxf(av0, av1);
      float e0 = __expf(av0 - mx), e1 = __expf(av1 - mx);
      float inv = 1.f / (e0 + e1);
      float wa0 = amp[p] * e0 * inv;
      float wa1 = amp[P_ + p] * e1 * inv;

      float wt[4]  = { (1.f - fx) * (1.f - fy), fx * (1.f - fy),
                       (1.f - fx) * fy,         fx * fy };
      int   idx[4] = { y0 * S_ + x0, y0 * S_ + x1, y1 * S_ + x0, y1 * S_ + x1 };

      float* im0 = imgR + (size_t)(b * C_ + 0) * S_ * S_;
      float* im1 = imgR + (size_t)(b * C_ + 1) * S_ * S_;
#pragma unroll
      for (int k = 0; k < 4; ++k) {
        atomicAdd(im0 + idx[k], wa0 * wt[k]);
        atomicAdd(im1 + idx[k], wa1 * wt[k]);
      }
    }
  }
}

// ------------------------------------------------------------------
// FFT machinery: 256-point radix-2 DIT in LDS, 128 threads per line
// ------------------------------------------------------------------
__device__ __forceinline__ void fft256(float2* buf, int tid, float dirsign) {
  for (int len = 2; len <= 256; len <<= 1) {
    int half = len >> 1;
    __syncthreads();
    int pos = tid & (half - 1);
    int i0 = ((tid & ~(half - 1)) << 1) + pos;
    int i1 = i0 + half;
    float ang = dirsign * 6.28318530718f * (float)pos / (float)len;
    float wr, wi;
    __sincosf(ang, &wi, &wr);
    float2 u = buf[i0], v = buf[i1];
    float vr = v.x * wr - v.y * wi;
    float vi = v.x * wi + v.y * wr;
    buf[i0] = make_float2(u.x + vr, u.y + vi);
    buf[i1] = make_float2(u.x - vr, u.y - vi);
  }
  __syncthreads();
}

__global__ void __launch_bounds__(128)
k_fft_rows_r2c(const float* __restrict__ src, float2* __restrict__ dst) {
  __shared__ float2 buf[256];
  int tid = threadIdx.x;
  size_t base = (size_t)blockIdx.x * 256;
#pragma unroll
  for (int e = 0; e < 2; ++e) {
    int i = tid + e * 128;
    int br = __brev((unsigned)i) >> 24;
    buf[i] = make_float2(src[base + br], 0.f);
  }
  fft256(buf, tid, -1.f);
#pragma unroll
  for (int e = 0; e < 2; ++e) {
    int i = tid + e * 128;
    dst[base + i] = buf[i];
  }
}

__global__ void __launch_bounds__(128)
k_fft_cols(float2* __restrict__ data, float dirsign, float scale) {
  __shared__ float2 buf[256];
  int tid = threadIdx.x;
  int img = blockIdx.x >> 8, x = blockIdx.x & 255;
  size_t base = (size_t)img * 65536 + x;
#pragma unroll
  for (int e = 0; e < 2; ++e) {
    int i = tid + e * 128;
    int br = __brev((unsigned)i) >> 24;
    buf[i] = data[base + (size_t)br * 256];
  }
  fft256(buf, tid, dirsign);
#pragma unroll
  for (int e = 0; e < 2; ++e) {
    int i = tid + e * 128;
    float2 vv = buf[i];
    data[base + (size_t)i * 256] = make_float2(vv.x * scale, vv.y * scale);
  }
}

__global__ void __launch_bounds__(128)
k_fft_rows_c2r(const float2* __restrict__ src, float* __restrict__ dst, float scale) {
  __shared__ float2 buf[256];
  int tid = threadIdx.x;
  size_t base = (size_t)blockIdx.x * 256;
#pragma unroll
  for (int e = 0; e < 2; ++e) {
    int i = tid + e * 128;
    int br = __brev((unsigned)i) >> 24;
    buf[i] = src[base + br];
  }
  fft256(buf, tid, 1.f);
#pragma unroll
  for (int e = 0; e < 2; ++e) {
    int i = tid + e * 128;
    dst[base + i] = buf[i].x * scale;
  }
}

// spectral combine: Fsum = (sum_c FF_c * F_bc) * shift-phase
__global__ void __launch_bounds__(256)
k_combine(const float2* __restrict__ Fa, float2* __restrict__ Fb,
          const float* __restrict__ Avec, const float* __restrict__ Bvec,
          const float* __restrict__ shift) {
  int idx = blockIdx.x * 256 + threadIdx.x;
  if (idx >= B_ * 65536) return;
  int b = idx >> 16, r = idx & 65535;
  int u = r >> 8, vx = r & 255;
  float cu = (float)(((u + 128) & 255) - 128);   // ifftshifted centered coords
  float cv = (float)(((vx + 128) & 255) - 128);
  float R = sqrtf(cu * cu + cv * cv);
  float F0 = Avec[0] * Avec[0] * __expf(-Bvec[0] * Bvec[0] * R);
  float F1 = Avec[1] * Avec[1] * __expf(-Bvec[1] * Bvec[1] * R);
  float2 a0 = Fa[(size_t)(b * 2 + 0) * 65536 + r];
  float2 a1 = Fa[(size_t)(b * 2 + 1) * 65536 + r];
  float sr = F0 * a0.x + F1 * a1.x;
  float si = F0 * a0.y + F1 * a1.y;
  float fu = ((u  < 128) ? (float)u  : (float)(u  - 256)) * (1.f / 256.f);
  float fv = ((vx < 128) ? (float)vx : (float)(vx - 256)) * (1.f / 256.f);
  float sx = shift[b * 2 + 0], sy = shift[b * 2 + 1];
  float ang = -6.28318530718f * (fv * sx + fu * sy);
  float pr, pi;
  __sincosf(ang, &pi, &pr);
  Fb[(size_t)b * 65536 + r] = make_float2(sr * pr - si * pi, sr * pi + si * pr);
}

__global__ void __launch_bounds__(256) k_zero(float* __restrict__ p, int n) {
  int i = blockIdx.x * 256 + threadIdx.x;
  if (i < n) p[i] = 0.f;
}

// ------------------------------------------------------------------
extern "C" void kernel_launch(void* const* d_in, const int* in_sizes, int n_in,
                              void* d_out, int out_size, void* d_ws, size_t ws_size,
                              hipStream_t stream) {
  (void)in_sizes; (void)n_in; (void)out_size; (void)ws_size;
  const float* z      = (const float*)d_in[0];
  const float* ori    = (const float*)d_in[1];
  const float* shift  = (const float*)d_in[2];
  const float* mpos   = (const float*)d_in[3];
  const float* Win    = (const float*)d_in[4];
  const float* Wl     = (const float*)d_in[5];
  const float* Wo1    = (const float*)d_in[6];
  const float* Wo2    = (const float*)d_in[7];
  const float* amp    = (const float*)d_in[8];
  const float* ampvar = (const float*)d_in[9];
  const float* Avec   = (const float*)d_in[10];
  const float* Bvec   = (const float*)d_in[11];

  float* out       = (float*)d_out;
  float* out_imgs  = out;                                     // [B,S,S]
  float* out_final = out + (size_t)B_ * S_ * S_;              // [B,P,3]
  float* out_disp  = out_final + (size_t)B_ * P_ * 3;         // [B,P,3]

  char* ws = (char*)d_ws;
  float*  imgR = (float*)ws;                                  // [B,C,S,S] real
  float2* cA   = (float2*)(ws + (size_t)B_ * C_ * S_ * S_ * 4);          // [B,C,S,S] cplx
  float2* cB   = (float2*)(ws + (size_t)B_ * C_ * S_ * S_ * (4 + 8));    // [B,S,S]   cplx

  const int nimg = B_ * C_ * S_ * S_;
  hipLaunchKernelGGL(k_zero, dim3((nimg + 255) / 256), dim3(256), 0, stream, imgR, nimg);

  hipLaunchKernelGGL(k_mlp_splat, dim3(512), dim3(256), 0, stream,
                     z, ori, mpos, Win, Wl, Wo1, Wo2, amp, ampvar,
                     imgR, out_final, out_disp);

  // forward 2D FFT per (b,c)
  hipLaunchKernelGGL(k_fft_rows_r2c, dim3(B_ * C_ * S_), dim3(128), 0, stream, imgR, cA);
  hipLaunchKernelGGL(k_fft_cols,     dim3(B_ * C_ * S_), dim3(128), 0, stream, cA, -1.f, 1.f);

  // filter * sum over classes * shift phase
  hipLaunchKernelGGL(k_combine, dim3(B_ * 65536 / 256), dim3(256), 0, stream,
                     cA, cB, Avec, Bvec, shift);

  // inverse 2D FFT per b, write real part to output
  hipLaunchKernelGGL(k_fft_cols,     dim3(B_ * S_), dim3(128), 0, stream, cB, 1.f, 1.f / 256.f);
  hipLaunchKernelGGL(k_fft_rows_c2r, dim3(B_ * S_), dim3(128), 0, stream, cB, out_imgs, 1.f / 256.f);
}
